// TransformerEncoder_50783693308405
// MI455X (gfx1250) — matrix operand order repaired and checked by reading, compiled, not run on hardware
//
#include <hip/hip_runtime.h>
#include <hip/hip_bf16.h>

typedef __attribute__((ext_vector_type(16))) _Float16 v16h;
typedef __attribute__((ext_vector_type(8)))  _Float16 v8h;
typedef __attribute__((ext_vector_type(8)))  float    v8f;

#define NTHREADS 256
#define HDIM 256
#define DFF  512
#define DMAX 8
#define KWORD 16
#define LAYERS 2

// ---------------------------------------------------------------------------
// Wave-level GEMM: D[16 x N] = A[16 x 256] * B[256 x N]  (+ optional acc)
// A: LDS f16, row-major 16x256 (rows >= MROWS are zero).
// B: global f16, row-major [256][N], 32B aligned.
// dst: LDS f32, only rows < MROWS stored (MROWS <= 8 -> only hb==0 lanes store).
// 8 waves split the N/16 tiles; K loop = 8 x v_wmma_f32_16x16x32_f16.
// ---------------------------------------------------------------------------
template <int MROWS, bool ACC>
__device__ __forceinline__ void wave_gemm(const _Float16* __restrict__ B, int N,
                                          const _Float16* __restrict__ A,
                                          float* __restrict__ dst, int ldD)
{
    const int lane = threadIdx.x & 31;
    const int wid  = threadIdx.x >> 5;
    const int hb   = lane >> 4;   // K half-select per ISA A/B layout
    const int l15  = lane & 15;
    const int tiles = N >> 4;
    for (int t = wid; t < tiles; t += 8) {
        const int n0 = t << 4;
        v8f c = {};
        if (ACC) {
            if (hb == 0) {
#pragma unroll
                for (int r = 0; r < 8; ++r)
                    if (r < MROWS) c[r] = dst[r * ldD + n0 + l15];
            }
        }
#pragma unroll
        for (int k0 = 0; k0 < 256; k0 += 32) {
            // A fragment: lane m=l15, K chunks [k0+hb*8 .. +7] and [k0+16+hb*8 .. +7]
            const _Float16* ap = A + l15 * 256 + k0 + hb * 8;
            v8h alo = *(const v8h*)(ap);
            v8h ahi = *(const v8h*)(ap + 16);
            v16h av;
#pragma unroll
            for (int e = 0; e < 8; ++e) { av[e] = alo[e]; av[e + 8] = ahi[e]; }
            // B fragment: lane n=l15, K chunks [k0+hb*8 .. +7] and [k0+16+hb*8 .. +7]
            v16h bv;
            const _Float16* bp = B + (size_t)(k0 + hb * 8) * N + n0 + l15;
#pragma unroll
            for (int e = 0; e < 8; ++e) { bv[e] = bp[(size_t)e * N]; bv[e + 8] = bp[(size_t)(16 + e) * N]; }
            c = __builtin_amdgcn_wmma_f32_16x16x32_f16(false, av, false, bv,
                                                       (short)0, c, false, false);
        }
        if (hb == 0) {
#pragma unroll
            for (int r = 0; r < 8; ++r)
                if (r < MROWS) dst[r * ldD + n0 + l15] = c[r];
        }
    }
}

// ---------------------------------------------------------------------------
// LayerNorm over 256-wide rows. One wave per row (rows <= 8).
// ---------------------------------------------------------------------------
__device__ __forceinline__ void ln_rows_f16(const float* __restrict__ src, int rows,
                                            const float* __restrict__ s,
                                            const float* __restrict__ b,
                                            _Float16* __restrict__ dstA)
{
    const int wid = threadIdx.x >> 5, lane = threadIdx.x & 31;
    if (wid < rows) {
        const float* r = src + wid * 256;
        float sum = 0.f, sq = 0.f;
#pragma unroll
        for (int j = 0; j < 8; ++j) { float v = r[lane + 32 * j]; sum += v; sq += v * v; }
#pragma unroll
        for (int o = 16; o > 0; o >>= 1) { sum += __shfl_xor(sum, o, 32); sq += __shfl_xor(sq, o, 32); }
        float mean = sum * (1.f / 256.f);
        float var  = sq * (1.f / 256.f) - mean * mean;
        float inv  = rsqrtf(var + 1e-6f);
#pragma unroll
        for (int j = 0; j < 8; ++j) {
            int c = lane + 32 * j;
            dstA[wid * 256 + c] = (_Float16)((r[c] - mean) * inv * s[c] + b[c]);
        }
    }
}

__device__ __forceinline__ void ln_rows_f32(const float* __restrict__ src, int rows,
                                            const float* __restrict__ s,
                                            const float* __restrict__ b,
                                            float* __restrict__ dst)
{
    const int wid = threadIdx.x >> 5, lane = threadIdx.x & 31;
    if (wid < rows) {
        const float* r = src + wid * 256;
        float sum = 0.f, sq = 0.f;
#pragma unroll
        for (int j = 0; j < 8; ++j) { float v = r[lane + 32 * j]; sum += v; sq += v * v; }
#pragma unroll
        for (int o = 16; o > 0; o >>= 1) { sum += __shfl_xor(sum, o, 32); sq += __shfl_xor(sq, o, 32); }
        float mean = sum * (1.f / 256.f);
        float var  = sq * (1.f / 256.f) - mean * mean;
        float inv  = rsqrtf(var + 1e-6f);
#pragma unroll
        for (int j = 0; j < 8; ++j) {
            int c = lane + 32 * j;
            dst[wid * 256 + c] = (r[c] - mean) * inv * s[c] + b[c];
        }
    }
}

// Single-row LN (decoder, seq len 1) on wave 0.
__device__ __forceinline__ void ln_row1(const float* __restrict__ src,
                                        const float* __restrict__ s,
                                        const float* __restrict__ b,
                                        float* __restrict__ dst)
{
    if (threadIdx.x < 32) {
        const int lane = threadIdx.x;
        float sum = 0.f, sq = 0.f;
#pragma unroll
        for (int j = 0; j < 8; ++j) { float v = src[lane + 32 * j]; sum += v; sq += v * v; }
#pragma unroll
        for (int o = 16; o > 0; o >>= 1) { sum += __shfl_xor(sum, o, 32); sq += __shfl_xor(sq, o, 32); }
        float mean = sum * (1.f / 256.f);
        float var  = sq * (1.f / 256.f) - mean * mean;
        float inv  = rsqrtf(var + 1e-6f);
#pragma unroll
        for (int j = 0; j < 8; ++j) {
            int c = lane + 32 * j;
            dst[c] = (src[c] - mean) * inv * s[c] + b[c];
        }
    }
}

// Re-stage an f32 LDS matrix (rows x 256 slice at colOff, leading dim ld) into
// the f16 A buffer, zero-padding rows >= rows.
__device__ __forceinline__ void stage_a(const float* __restrict__ src, int ld, int colOff,
                                        int rows, _Float16* __restrict__ dstA)
{
    for (int i = threadIdx.x; i < 4096; i += NTHREADS) {
        int m = i >> 8, c = i & 255;
        dstA[i] = (m < rows) ? (_Float16)src[m * ld + colOff + c] : (_Float16)0.f;
    }
}

// ---------------------------------------------------------------------------
// Kernel 1: f32 -> f16 weight conversion
// ---------------------------------------------------------------------------
__global__ void cvt_f16_kernel(const float* __restrict__ src, _Float16* __restrict__ dst, int n)
{
    int i = blockIdx.x * NTHREADS + threadIdx.x;
    if (i < n) dst[i] = (_Float16)src[i];
}

// ---------------------------------------------------------------------------
// Kernel 2: leaf embeddings  node_h[n,h] = sum_k x_word[n,k] * E_bu[h, idx[n,k]]
// ---------------------------------------------------------------------------
__global__ void leaf_kernel(const float* __restrict__ xw, const int* __restrict__ xi,
                            const float* __restrict__ E, int V,
                            float* __restrict__ node_h)
{
    const int n = blockIdx.x;
    const int h = threadIdx.x;
    __shared__ float w[KWORD];
    __shared__ int   id[KWORD];
    if (threadIdx.x < KWORD) {
        w[threadIdx.x]  = xw[(size_t)n * KWORD + threadIdx.x];
        id[threadIdx.x] = xi[(size_t)n * KWORD + threadIdx.x];
    }
    __syncthreads();
    float a = 0.f;
#pragma unroll
    for (int k = 0; k < KWORD; ++k) a = fmaf(w[k], E[(size_t)h * V + id[k]], a);
    node_h[(size_t)n * HDIM + h] = a;
}

// ---------------------------------------------------------------------------
// Kernel 3: sequential tree scan — one persistent workgroup (8 wave32 waves)
// ---------------------------------------------------------------------------
__global__ void __launch_bounds__(NTHREADS, 1)
scan_kernel(const float* __restrict__ xw, const int* __restrict__ xi,
            const int* __restrict__ tree,
            const float* __restrict__ Ebu, int V, int NL, int NP,
            float* __restrict__ node_h,
            const _Float16* __restrict__ encW16, const float* __restrict__ enc_ab,
            const _Float16* __restrict__ encF1,  const float* __restrict__ enc_fb1,
            const _Float16* __restrict__ encF2,  const float* __restrict__ enc_fb2,
            const float* __restrict__ enc_ln_s,  const float* __restrict__ enc_ln_b,
            const float* __restrict__ enc_fin_s, const float* __restrict__ enc_fin_b,
            const _Float16* __restrict__ decW16, const float* __restrict__ dec_ab,
            const _Float16* __restrict__ decF1,  const float* __restrict__ dec_fb1,
            const _Float16* __restrict__ decF2,  const float* __restrict__ dec_fb2,
            const float* __restrict__ dec_ln_s,  const float* __restrict__ dec_ln_b,
            const float* __restrict__ dec_fin_s, const float* __restrict__ dec_fin_b)
{
    __shared__ __align__(32) _Float16 sA[16 * 256];  // WMMA A staging (rows>=used are 0)
    __shared__ float sT0[8 * 512];                   // Q / FF1 / generic row scratch
    __shared__ float sK[8 * 256];                    // K, then attention output
    __shared__ float sV[8 * 256];
    __shared__ float sX[8 * 256];                    // encoder residual stream
    __shared__ float sMem[8 * 256];                  // encoder output
    __shared__ float sS[512];                        // attention scores / probs
    __shared__ float sDx[256];                       // decoder residual
    __shared__ float sDy[256];                       // decoder LN / attn-out scratch
    __shared__ float sDv[512];                       // decoder intermediate (Q row / FF2 out)
    __shared__ int   sCh[DMAX];
    __shared__ int   sMask[DMAX];
    __shared__ float sWds[KWORD];
    __shared__ int   sIdx[KWORD];

    const float iscale = 0.17677669529663687f;  // 1/sqrt(32)

    for (int i = threadIdx.x; i < 4096; i += NTHREADS) sA[i] = (_Float16)0.f;
    __syncthreads();

    for (int step = 0; step < NP; ++step) {
        if (threadIdx.x < DMAX) {
            int c = tree[step * DMAX + threadIdx.x];
            sCh[threadIdx.x] = c;
            sMask[threadIdx.x] = (c >= 0);
        }
        if (threadIdx.x < KWORD) {
            sWds[threadIdx.x] = xw[(size_t)(NL + step) * KWORD + threadIdx.x];
            sIdx[threadIdx.x] = xi[(size_t)(NL + step) * KWORD + threadIdx.x];
        }
        __syncthreads();

        // gather children into sX; decoder input embedding into sDx
        for (int idx = threadIdx.x; idx < 2048; idx += NTHREADS) {
            int j = idx >> 8, col = idx & 255;
            int c = sCh[j]; if (c < 0) c = 0;
            sX[idx] = node_h[(size_t)c * HDIM + col];
        }
        {
            const int h = threadIdx.x;
            float a = 0.f;
#pragma unroll
            for (int k = 0; k < KWORD; ++k) a = fmaf(sWds[k], Ebu[(size_t)h * V + sIdx[k]], a);
            sDx[h] = a;
        }
        __syncthreads();

        // ===================== encoder (2 layers, seq len 8) =====================
        for (int l = 0; l < LAYERS; ++l) {
            // ---- MHA ----
            ln_rows_f16(sX, 8, enc_ln_s + (l * 2 + 0) * 256, enc_ln_b + (l * 2 + 0) * 256, sA);
            __syncthreads();
            wave_gemm<8, false>(encW16 + (size_t)(l * 4 + 0) * 65536, 256, sA, sT0, 256); // Q
            wave_gemm<8, false>(encW16 + (size_t)(l * 4 + 1) * 65536, 256, sA, sK,  256); // K
            wave_gemm<8, false>(encW16 + (size_t)(l * 4 + 2) * 65536, 256, sA, sV,  256); // V
            __syncthreads();
            for (int idx = threadIdx.x; idx < 2048; idx += NTHREADS) {
                int col = idx & 255;
                sT0[idx] += enc_ab[(l * 4 + 0) * 256 + col];
                sK[idx]  += enc_ab[(l * 4 + 1) * 256 + col];
                sV[idx]  += enc_ab[(l * 4 + 2) * 256 + col];
            }
            __syncthreads();
            // scores s[h][q][k]
            for (int idx = threadIdx.x; idx < 512; idx += NTHREADS) {
                int h = idx >> 6, q = (idx >> 3) & 7, k = idx & 7;
                const float* qp = sT0 + q * 256 + h * 32;
                const float* kp = sK  + k * 256 + h * 32;
                float a = 0.f;
#pragma unroll
                for (int d = 0; d < 32; ++d) a = fmaf(qp[d], kp[d], a);
                sS[idx] = a * iscale;
            }
            __syncthreads();
            // masked softmax over k
            if (threadIdx.x < 64) {
                float* row = sS + threadIdx.x * 8;
                float mx = -1e30f;
#pragma unroll
                for (int k = 0; k < DMAX; ++k) { float v = sMask[k] ? row[k] : -1e9f; mx = fmaxf(mx, v); }
                float e[DMAX]; float sum = 0.f;
#pragma unroll
                for (int k = 0; k < DMAX; ++k) { float v = sMask[k] ? __expf(row[k] - mx) : 0.f; e[k] = v; sum += v; }
                float r = 1.f / sum;
#pragma unroll
                for (int k = 0; k < DMAX; ++k) row[k] = e[k] * r;
            }
            __syncthreads();
            // o[q][col] = sum_k a[h][q][k] * V[k][col]  (K dead -> write into sK)
            {
                const int col = threadIdx.x;
                const int h = col >> 5;
#pragma unroll
                for (int q = 0; q < 8; ++q) {
                    float a = 0.f;
#pragma unroll
                    for (int k = 0; k < DMAX; ++k) a = fmaf(sS[h * 64 + q * 8 + k], sV[k * 256 + col], a);
                    sK[q * 256 + col] = a;
                }
            }
            __syncthreads();
            stage_a(sK, 256, 0, 8, sA);
            __syncthreads();
            wave_gemm<8, false>(encW16 + (size_t)(l * 4 + 3) * 65536, 256, sA, sT0, 256); // O proj
            __syncthreads();
            for (int idx = threadIdx.x; idx < 2048; idx += NTHREADS)
                sX[idx] += sT0[idx] + enc_ab[(l * 4 + 3) * 256 + (idx & 255)];
            __syncthreads();

            // ---- FF ----
            ln_rows_f16(sX, 8, enc_ln_s + (l * 2 + 1) * 256, enc_ln_b + (l * 2 + 1) * 256, sA);
            __syncthreads();
            wave_gemm<8, false>(encF1 + (size_t)l * 131072, 512, sA, sT0, 512);
            __syncthreads();
            for (int idx = threadIdx.x; idx < 4096; idx += NTHREADS)
                sT0[idx] = fmaxf(sT0[idx] + enc_fb1[l * 512 + (idx & 511)], 0.f);
            __syncthreads();
            stage_a(sT0, 512, 0, 8, sA);            // K-half 0
            __syncthreads();
            wave_gemm<8, false>(encF2 + (size_t)l * 131072, 256, sA, sK, 256);
            __syncthreads();
            stage_a(sT0, 512, 256, 8, sA);          // K-half 1
            __syncthreads();
            wave_gemm<8, true>(encF2 + (size_t)l * 131072 + 65536, 256, sA, sK, 256);
            __syncthreads();
            for (int idx = threadIdx.x; idx < 2048; idx += NTHREADS)
                sX[idx] += sK[idx] + enc_fb2[l * 256 + (idx & 255)];
            __syncthreads();
        }
        ln_rows_f32(sX, 8, enc_fin_s, enc_fin_b, sMem);
        __syncthreads();

        // ===================== decoder (2 layers, query len 1) =====================
        for (int l = 0; l < LAYERS; ++l) {
            const int base = l * 2;
            // ---- self-attn (len 1 => softmax == 1 => out = (y@Wv+bv)@Wo+bo) ----
            ln_row1(sDx, dec_ln_s + (l * 3 + 0) * 256, dec_ln_b + (l * 3 + 0) * 256, sDy);
            __syncthreads();
            stage_a(sDy, 256, 0, 1, sA);
            __syncthreads();
            wave_gemm<1, false>(decW16 + (size_t)((base + 0) * 4 + 2) * 65536, 256, sA, sT0, 256);
            __syncthreads();
            sT0[threadIdx.x] += dec_ab[((base + 0) * 4 + 2) * 256 + threadIdx.x];
            __syncthreads();
            stage_a(sT0, 256, 0, 1, sA);
            __syncthreads();
            wave_gemm<1, false>(decW16 + (size_t)((base + 0) * 4 + 3) * 65536, 256, sA, sT0, 256);
            __syncthreads();
            sDx[threadIdx.x] += sT0[threadIdx.x] + dec_ab[((base + 0) * 4 + 3) * 256 + threadIdx.x];
            __syncthreads();

            // ---- cross-attn to encoder memory ----
            ln_row1(sDx, dec_ln_s + (l * 3 + 1) * 256, dec_ln_b + (l * 3 + 1) * 256, sDy);
            __syncthreads();
            stage_a(sDy, 256, 0, 1, sA);
            __syncthreads();
            wave_gemm<1, false>(decW16 + (size_t)((base + 1) * 4 + 0) * 65536, 256, sA, sDv, 256); // Q
            __syncthreads();
            sDv[threadIdx.x] += dec_ab[((base + 1) * 4 + 0) * 256 + threadIdx.x];
            stage_a(sMem, 256, 0, 8, sA);
            __syncthreads();
            wave_gemm<8, false>(decW16 + (size_t)((base + 1) * 4 + 1) * 65536, 256, sA, sK, 256);
            wave_gemm<8, false>(decW16 + (size_t)((base + 1) * 4 + 2) * 65536, 256, sA, sV, 256);
            __syncthreads();
            for (int idx = threadIdx.x; idx < 2048; idx += NTHREADS) {
                int col = idx & 255;
                sK[idx] += dec_ab[((base + 1) * 4 + 1) * 256 + col];
                sV[idx] += dec_ab[((base + 1) * 4 + 2) * 256 + col];
            }
            __syncthreads();
            if (threadIdx.x < 64) {
                int h = threadIdx.x >> 3, k = threadIdx.x & 7;
                const float* qp = sDv + h * 32;
                const float* kp = sK + k * 256 + h * 32;
                float a = 0.f;
#pragma unroll
                for (int d = 0; d < 32; ++d) a = fmaf(qp[d], kp[d], a);
                sS[h * 8 + k] = a * iscale;
            }
            __syncthreads();
            if (threadIdx.x < 8) {
                float* row = sS + threadIdx.x * 8;
                float mx = -1e30f;
#pragma unroll
                for (int k = 0; k < DMAX; ++k) { float v = sMask[k] ? row[k] : -1e9f; mx = fmaxf(mx, v); }
                float sum = 0.f;
#pragma unroll
                for (int k = 0; k < DMAX; ++k) { float e = sMask[k] ? __expf(row[k] - mx) : 0.f; row[k] = e; sum += e; }
                float r = 1.f / sum;
#pragma unroll
                for (int k = 0; k < DMAX; ++k) row[k] *= r;
            }
            __syncthreads();
            {
                const int col = threadIdx.x;
                const int h = col >> 5;
                float a = 0.f;
#pragma unroll
                for (int k = 0; k < DMAX; ++k) a = fmaf(sS[h * 8 + k], sV[k * 256 + col], a);
                sDy[col] = a;
            }
            __syncthreads();
            stage_a(sDy, 256, 0, 1, sA);
            __syncthreads();
            wave_gemm<1, false>(decW16 + (size_t)((base + 1) * 4 + 3) * 65536, 256, sA, sT0, 256);
            __syncthreads();
            sDx[threadIdx.x] += sT0[threadIdx.x] + dec_ab[((base + 1) * 4 + 3) * 256 + threadIdx.x];
            __syncthreads();

            // ---- FF ----
            ln_row1(sDx, dec_ln_s + (l * 3 + 2) * 256, dec_ln_b + (l * 3 + 2) * 256, sDy);
            __syncthreads();
            stage_a(sDy, 256, 0, 1, sA);
            __syncthreads();
            wave_gemm<1, false>(decF1 + (size_t)l * 131072, 512, sA, sT0, 512);
            __syncthreads();
            for (int idx = threadIdx.x; idx < 512; idx += NTHREADS)
                sT0[idx] = fmaxf(sT0[idx] + dec_fb1[l * 512 + idx], 0.f);
            __syncthreads();
            stage_a(sT0, 512, 0, 1, sA);            // K-half 0
            __syncthreads();
            wave_gemm<1, false>(decF2 + (size_t)l * 131072, 256, sA, sDv, 256);
            __syncthreads();
            stage_a(sT0, 512, 256, 1, sA);          // K-half 1
            __syncthreads();
            wave_gemm<1, true>(decF2 + (size_t)l * 131072 + 65536, 256, sA, sDv, 256);
            __syncthreads();
            sDx[threadIdx.x] += sDv[threadIdx.x] + dec_fb2[l * 256 + threadIdx.x];
            __syncthreads();
        }
        ln_row1(sDx, dec_fin_s, dec_fin_b, sDy);
        __syncthreads();
        node_h[(size_t)(NL + step) * HDIM + threadIdx.x] = sDy[threadIdx.x];
        __threadfence();
        __syncthreads();
    }
}

// ---------------------------------------------------------------------------
// Kernel 4: readout softmax over C classes of the root node
// ---------------------------------------------------------------------------
__global__ void out_kernel(const float* __restrict__ node_h, int rootIdx,
                           const float* __restrict__ Wout, const float* __restrict__ bout,
                           int C, float* __restrict__ out)
{
    __shared__ float logits[8];
    const int wid = threadIdx.x >> 5, lane = threadIdx.x & 31;
    if (wid < C) {
        const float* r = node_h + (size_t)rootIdx * HDIM;
        const float* w = Wout + (size_t)wid * HDIM;
        float a = 0.f;
        for (int j = lane; j < HDIM; j += 32) a = fmaf(r[j], w[j], a);
#pragma unroll
        for (int o = 16; o > 0; o >>= 1) a += __shfl_xor(a, o, 32);
        if (lane == 0) logits[wid] = a + bout[wid];
    }
    __syncthreads();
    if (threadIdx.x == 0) {
        float mx = -1e30f;
        for (int c = 0; c < C; ++c) mx = fmaxf(mx, logits[c]);
        float s = 0.f;
        for (int c = 0; c < C; ++c) { float e = __expf(logits[c] - mx); logits[c] = e; s += e; }
        for (int c = 0; c < C; ++c) out[c] = logits[c] / s;
    }
}

// ---------------------------------------------------------------------------
extern "C" void kernel_launch(void* const* d_in, const int* in_sizes, int n_in,
                              void* d_out, int out_size, void* d_ws, size_t ws_size,
                              hipStream_t stream)
{
    const float* x_word     = (const float*)d_in[0];
    const int*   x_index    = (const int*)  d_in[1];
    const int*   tree       = (const int*)  d_in[2];
    const float* E_bu       = (const float*)d_in[3];
    const float* W_out      = (const float*)d_in[4];
    const float* b_out      = (const float*)d_in[5];
    const float* enc_attn_w = (const float*)d_in[6];
    const float* enc_attn_b = (const float*)d_in[7];
    const float* enc_ff_w1  = (const float*)d_in[8];
    const float* enc_ff_b1  = (const float*)d_in[9];
    const float* enc_ff_w2  = (const float*)d_in[10];
    const float* enc_ff_b2  = (const float*)d_in[11];
    const float* enc_ln_s   = (const float*)d_in[12];
    const float* enc_ln_b   = (const float*)d_in[13];
    const float* enc_fin_s  = (const float*)d_in[14];
    const float* enc_fin_b  = (const float*)d_in[15];
    const float* dec_attn_w = (const float*)d_in[16];
    const float* dec_attn_b = (const float*)d_in[17];
    const float* dec_ff_w1  = (const float*)d_in[18];
    const float* dec_ff_b1  = (const float*)d_in[19];
    const float* dec_ff_w2  = (const float*)d_in[20];
    const float* dec_ff_b2  = (const float*)d_in[21];
    const float* dec_ln_s   = (const float*)d_in[22];
    const float* dec_ln_b   = (const float*)d_in[23];
    const float* dec_fin_s  = (const float*)d_in[24];
    const float* dec_fin_b  = (const float*)d_in[25];

    const int Ntot = in_sizes[0] / KWORD;     // 16384
    const int NP   = in_sizes[2] / DMAX;      // 4096
    const int NL   = Ntot - NP;               // 12288
    const int V    = in_sizes[3] / HDIM;      // 50000
    const int C    = in_sizes[4] / HDIM;      // 4

    // ---- workspace carve (all offsets keep 32B alignment) ----
    char* ws = (char*)d_ws;
    float*     node_h = (float*)ws;     ws += (size_t)Ntot * HDIM * sizeof(float);
    _Float16*  encW16 = (_Float16*)ws;  ws += (size_t)LAYERS * 4 * HDIM * HDIM * sizeof(_Float16);
    _Float16*  encF1  = (_Float16*)ws;  ws += (size_t)LAYERS * HDIM * DFF * sizeof(_Float16);
    _Float16*  encF2  = (_Float16*)ws;  ws += (size_t)LAYERS * DFF * HDIM * sizeof(_Float16);
    _Float16*  decW16 = (_Float16*)ws;  ws += (size_t)LAYERS * 2 * 4 * HDIM * HDIM * sizeof(_Float16);
    _Float16*  decF1  = (_Float16*)ws;  ws += (size_t)LAYERS * HDIM * DFF * sizeof(_Float16);
    _Float16*  decF2  = (_Float16*)ws;  ws += (size_t)LAYERS * DFF * HDIM * sizeof(_Float16);

    // ---- weight conversion (once per launch; resident in L2 afterwards) ----
    int n1 = LAYERS * 4 * HDIM * HDIM;
    cvt_f16_kernel<<<(n1 + NTHREADS - 1) / NTHREADS, NTHREADS, 0, stream>>>(enc_attn_w, encW16, n1);
    int n2 = LAYERS * HDIM * DFF;
    cvt_f16_kernel<<<(n2 + NTHREADS - 1) / NTHREADS, NTHREADS, 0, stream>>>(enc_ff_w1, encF1, n2);
    int n3 = LAYERS * DFF * HDIM;
    cvt_f16_kernel<<<(n3 + NTHREADS - 1) / NTHREADS, NTHREADS, 0, stream>>>(enc_ff_w2, encF2, n3);
    int n4 = LAYERS * 2 * 4 * HDIM * HDIM;
    cvt_f16_kernel<<<(n4 + NTHREADS - 1) / NTHREADS, NTHREADS, 0, stream>>>(dec_attn_w, decW16, n4);
    cvt_f16_kernel<<<(n2 + NTHREADS - 1) / NTHREADS, NTHREADS, 0, stream>>>(dec_ff_w1, decF1, n2);
    cvt_f16_kernel<<<(n3 + NTHREADS - 1) / NTHREADS, NTHREADS, 0, stream>>>(dec_ff_w2, decF2, n3);

    // ---- leaf embeddings (massively parallel) ----
    leaf_kernel<<<NL, NTHREADS, 0, stream>>>(x_word, x_index, E_bu, V, node_h);

    // ---- sequential tree scan: single persistent workgroup ----
    scan_kernel<<<1, NTHREADS, 0, stream>>>(
        x_word, x_index, tree, E_bu, V, NL, NP, node_h,
        encW16, enc_attn_b, encF1, enc_ff_b1, encF2, enc_ff_b2,
        enc_ln_s, enc_ln_b, enc_fin_s, enc_fin_b,
        decW16, dec_attn_b, decF1, dec_ff_b1, decF2, dec_ff_b2,
        dec_ln_s, dec_ln_b, dec_fin_s, dec_fin_b);

    // ---- readout ----
    out_kernel<<<1, 128, 0, stream>>>(node_h, Ntot - 1, W_out, b_out, C, (float*)d_out);
}
